// DrugInteractionAttention_48799418417313
// MI455X (gfx1250) — matrix operand order repaired and checked by reading, not compiled, not run on hardware
//
#include <hip/hip_runtime.h>
#include <hip/hip_bf16.h>
#include <stdint.h>

// ---------------------------------------------------------------------------
// DrugInteractionAttention, seq-len == 1  =>  attention == V/O projections.
//   v_i    = X_i @ Wv[i] + bv[i]                      (bf16 panels in ws)
//   att    = v0 @ Wo[0] + bo[0] + v1 @ Wo[1] + bo[1]  (f32 panel in ws)
//   comb   = [ v2@Wo[2]+bo[2] | v3@Wo[3]+bo[3] ]      (bf16 [B,2048] in ws)
//   fused  = att * sigmoid(comb @ Wg + bg)            (f32 -> d_out)
//   logits = comb @ Wc + bc                           (f32 -> d_out + B*D)
// GEMMs: v_wmma_f32_16x16x32_bf16, f32 accumulate, double-buffered LDS,
// async global->LDS copies for bf16 A-panels, sched_barrier-fenced pipeline.
// ---------------------------------------------------------------------------

typedef __attribute__((ext_vector_type(16))) __bf16 bf16x16;
typedef __attribute__((ext_vector_type(8)))  __bf16 bf16x8;
typedef __attribute__((ext_vector_type(4)))  __bf16 bf16x4;
typedef __attribute__((ext_vector_type(8)))  float  f32x8;
typedef __attribute__((ext_vector_type(4)))  float  f32x4;
typedef __attribute__((ext_vector_type(4)))  int    i32x4;

#define TM 128
#define TN 128
#define TK 32
#define LDB 1024              // all weight panels are [K,1024] row-major
#define AS_STRIDE (TK + 8)    // bf16 units; 40*2B = 80B rows (16B aligned)
#define BS_STRIDE (TN + 8)    // bf16 units; 136*2B = 272B rows (16B aligned)

#define EPI_STORE_BF16 0
#define EPI_F32_INIT   1
#define EPI_F32_ACCUM  2
#define EPI_GATE_FUSE  3

#if __has_builtin(__builtin_amdgcn_global_load_async_to_lds_b128) && \
    __has_builtin(__builtin_amdgcn_s_wait_asynccnt)
#define DIA_ASYNC_LDS 1
#else
#define DIA_ASYNC_LDS 0
#endif

#if __has_builtin(__builtin_amdgcn_sched_barrier)
#define DIA_SCHED_FENCE() __builtin_amdgcn_sched_barrier(0)
#else
#define DIA_SCHED_FENCE()
#endif

#if DIA_ASYNC_LDS
typedef __attribute__((address_space(1))) i32x4 g_i32x4;   // global int4
typedef __attribute__((address_space(3))) i32x4 l_i32x4;   // LDS int4
__device__ __forceinline__ void dia_async_copy16(const void* g, void* l) {
    // generic LDS pointers carry the LDS byte offset in their low 32 bits,
    // generic global pointers equal their AS1 value.
    __builtin_amdgcn_global_load_async_to_lds_b128(
        (g_i32x4*)(uintptr_t)g, (l_i32x4*)(uintptr_t)l, 0, 0);
}
#endif

union frag_u { bf16x16 v; bf16x8 h[2]; };

template <int MODE, bool ABF16, int LDA>
__global__ __launch_bounds__(256)
void dia_gemm(const void* __restrict__ Aptr,
              const float* __restrict__ Bw,            // f32 weights [K, LDB]
              const float* __restrict__ bias,          // [N]
              void* __restrict__ Cptr, int ldc, int c_col_off,
              const float* __restrict__ attsum,        // used by EPI_GATE_FUSE
              int K)
{
    __shared__ __bf16 As[2][TM * AS_STRIDE];
    __shared__ __bf16 Bs[2][TK * BS_STRIDE];

    const int tid   = threadIdx.x;
    const int m0    = blockIdx.y * TM;
    const int n0    = blockIdx.x * TN;
    const int wave  = tid >> 5;
    const int lane  = tid & 31;
    const int wm    = wave >> 2;          // 0..1 : 64-row strip
    const int wn    = wave & 3;           // 0..3 : 32-col strip
    const int lrow  = lane & 15;
    const int lhalf = lane >> 4;

    f32x8 acc[4][2];
#pragma unroll
    for (int i = 0; i < 4; ++i)
#pragma unroll
        for (int j = 0; j < 2; ++j) {
            f32x8 z = {0.f, 0.f, 0.f, 0.f, 0.f, 0.f, 0.f, 0.f};
            acc[i][j] = z;
        }

    // ---- per-thread global pointers, advanced by constant stride per step -
    const int arF = tid >> 3, aqF = tid & 7;     // f32 A path: 4x float4
    const int arH = tid >> 2, aqH = tid & 3;     // bf16 A path: 2x bf16x8
    const int bkr = tid >> 5, bq  = tid & 31;    // B: 4x float4

    const float*  aF[4];
    const __bf16* aH[2];
    const float*  bP[4];
    if constexpr (!ABF16) {
#pragma unroll
        for (int i = 0; i < 4; ++i)
            aF[i] = (const float*)Aptr + (size_t)(m0 + arF + 32 * i) * LDA + aqF * 4;
    } else {
#pragma unroll
        for (int i = 0; i < 2; ++i)
            aH[i] = (const __bf16*)Aptr + (size_t)(m0 + arH + 64 * i) * LDA + aqH * 8;
    }
#pragma unroll
    for (int i = 0; i < 4; ++i)
        bP[i] = Bw + (size_t)(bkr + 8 * i) * LDB + n0 + bq * 4;

    f32x4  ra[4];    // staged A (f32 path)
    f32x4  rb[4];    // staged B (always f32 weights)

    // ---- A tile fetch: async LDS copy (bf16) / reg stage (f32) ------------
    auto fetch_a = [&](int dbuf) {
        if constexpr (ABF16) {
#if DIA_ASYNC_LDS
#pragma unroll
            for (int i = 0; i < 2; ++i) {
                dia_async_copy16(aH[i], &As[dbuf][(arH + 64 * i) * AS_STRIDE + aqH * 8]);
                aH[i] += TK;
            }
#else
            (void)dbuf;
#pragma unroll
            for (int i = 0; i < 2; ++i) {
                ra[i] = *(const f32x4*)(const void*)aH[i];   // 8 bf16 = 16B raw
                aH[i] += TK;
            }
#endif
        } else {
            (void)dbuf;
#pragma unroll
            for (int i = 0; i < 4; ++i) {
                ra[i] = *(const f32x4*)aF[i];
                aF[i] += TK;
            }
        }
    };
    auto commit_a = [&](int dbuf) {                  // reg paths only
        if constexpr (ABF16) {
#if !DIA_ASYNC_LDS
#pragma unroll
            for (int i = 0; i < 2; ++i)
                *(f32x4*)(void*)&As[dbuf][(arH + 64 * i) * AS_STRIDE + aqH * 8] = ra[i];
#else
            (void)dbuf;
#endif
        } else {
#pragma unroll
            for (int i = 0; i < 4; ++i) {
                bf16x4 pk;
                pk[0] = (__bf16)ra[i][0]; pk[1] = (__bf16)ra[i][1];
                pk[2] = (__bf16)ra[i][2]; pk[3] = (__bf16)ra[i][3];
                *(bf16x4*)&As[dbuf][(arF + 32 * i) * AS_STRIDE + aqF * 4] = pk;
            }
        }
    };
    // ---- B tile: global fetch to regs, cvt+commit to LDS ------------------
    auto fetch_b = [&]() {
#pragma unroll
        for (int i = 0; i < 4; ++i) {
            rb[i] = *(const f32x4*)bP[i];
            bP[i] += (size_t)TK * LDB;
        }
    };
    auto commit_b = [&](int dbuf) {
#pragma unroll
        for (int i = 0; i < 4; ++i) {
            bf16x4 pk;
            pk[0] = (__bf16)rb[i][0]; pk[1] = (__bf16)rb[i][1];
            pk[2] = (__bf16)rb[i][2]; pk[3] = (__bf16)rb[i][3];
            *(bf16x4*)&Bs[dbuf][(bkr + 8 * i) * BS_STRIDE + bq * 4] = pk;
        }
    };
    auto finish_tile = [&]() {
#if DIA_ASYNC_LDS
        if constexpr (ABF16) __builtin_amdgcn_s_wait_asynccnt(0);
#endif
        __syncthreads();
    };
    // ---- fragment loads + 8 WMMAs per K-step ------------------------------
    // A 16x32 bf16: lane -> M=lane&15; elems 0..7 => K=8*half+e, 8..15 => 16+8*half+(e-8)
    // B 32x16 bf16: lane -> N=lane&15; elems 0..7 => K=8*half+e, 8..15 => 16+8*half+(e-8)
    auto compute = [&](int cbuf) {
        frag_u af[4], bfv[2];
#pragma unroll
        for (int fm = 0; fm < 4; ++fm) {
            const __bf16* p = &As[cbuf][(wm * 64 + fm * 16 + lrow) * AS_STRIDE + lhalf * 8];
            af[fm].h[0] = *(const bf16x8*)p;
            af[fm].h[1] = *(const bf16x8*)(p + 16);
        }
#pragma unroll
        for (int fn = 0; fn < 2; ++fn) {
            const __bf16* p = &Bs[cbuf][(lhalf * 8) * BS_STRIDE + wn * 32 + fn * 16 + lrow];
#pragma unroll
            for (int e = 0; e < 8; ++e) {
                bfv[fn].v[e]     = p[e * BS_STRIDE];
                bfv[fn].v[8 + e] = p[(16 + e) * BS_STRIDE];
            }
        }
#pragma unroll
        for (int fm = 0; fm < 4; ++fm)
#pragma unroll
            for (int fn = 0; fn < 2; ++fn)
                acc[fm][fn] = __builtin_amdgcn_wmma_f32_16x16x32_bf16(
                    false, af[fm].v, false, bfv[fn].v,
                    (short)0, acc[fm][fn], false, false);
    };

    // ---- software-pipelined main loop (double-buffered LDS) ---------------
    fetch_a(0);
    fetch_b();
    commit_a(0);
    commit_b(0);
    finish_tile();

    int buf = 0;
    for (int k0 = TK; k0 < K; k0 += TK) {
        int nbuf = buf ^ 1;
        fetch_a(nbuf);           // 8 global loads / async LDS copies in flight
        fetch_b();
        DIA_SCHED_FENCE();       // keep loads issued above the WMMA block
        compute(buf);            // WMMA on current tile hides the latency
        DIA_SCHED_FENCE();       // keep cvt+stores (and their wait) below it
        commit_a(nbuf);
        commit_b(nbuf);
        finish_tile();
        buf = nbuf;
    }
    compute(buf);

    // ---- epilogue: C/D layout lane->N=lane&15, vgpr v -> M = 8*half+v -----
#pragma unroll
    for (int fm = 0; fm < 4; ++fm) {
#pragma unroll
        for (int fn = 0; fn < 2; ++fn) {
            int n = n0 + wn * 32 + fn * 16 + lrow;
            float bn = bias[n];
#pragma unroll
            for (int v = 0; v < 8; ++v) {
                int m = m0 + wm * 64 + fm * 16 + lhalf * 8 + v;
                float val = acc[fm][fn][v] + bn;
                size_t off = (size_t)m * ldc + c_col_off + n;
                if constexpr (MODE == EPI_STORE_BF16) {
                    ((__bf16*)Cptr)[off] = (__bf16)val;
                } else if constexpr (MODE == EPI_F32_INIT) {
                    ((float*)Cptr)[off] = val;
                } else if constexpr (MODE == EPI_F32_ACCUM) {
                    ((float*)Cptr)[off] += val;
                } else {  // EPI_GATE_FUSE: out = att * sigmoid(val)
                    float g = 1.f / (1.f + __expf(-val));
                    ((float*)Cptr)[off] = attsum[(size_t)m * 1024 + n] * g;
                }
            }
        }
    }
}

// logits = comb([B,2048] bf16) @ Wc([2048,3] f32) + bc ; one wave per row.
__global__ __launch_bounds__(256)
void dia_logits(const __bf16* __restrict__ comb, const float* __restrict__ Wc,
                const float* __restrict__ bc, float* __restrict__ out, int rows)
{
    int wave = threadIdx.x >> 5;
    int lane = threadIdx.x & 31;
    int row  = blockIdx.x * 8 + wave;
    if (row >= rows) return;
    const __bf16* cr = comb + (size_t)row * 2048;
    float s0 = 0.f, s1 = 0.f, s2 = 0.f;
    for (int k = lane; k < 2048; k += 32) {
        float x = (float)cr[k];
        s0 += x * Wc[k * 3 + 0];
        s1 += x * Wc[k * 3 + 1];
        s2 += x * Wc[k * 3 + 2];
    }
#pragma unroll
    for (int off = 16; off > 0; off >>= 1) {
        s0 += __shfl_down(s0, off, 32);
        s1 += __shfl_down(s1, off, 32);
        s2 += __shfl_down(s2, off, 32);
    }
    if (lane == 0) {
        float* o = out + (size_t)row * 3;
        o[0] = s0 + bc[0]; o[1] = s1 + bc[1]; o[2] = s2 + bc[2];
    }
}

extern "C" void kernel_launch(void* const* d_in, const int* in_sizes, int n_in,
                              void* d_out, int out_size, void* d_ws, size_t ws_size,
                              hipStream_t stream)
{
    (void)in_sizes; (void)n_in; (void)out_size; (void)ws_size;
    const int D = 1024, B = 16384;
    const size_t DD = (size_t)D * D;

    const float* drug_a = (const float*)d_in[0];
    const float* drug_b = (const float*)d_in[1];
    // d_in[2..5] = Wq, bq, Wk, bk : dead code (softmax over length-1 axis == 1)
    const float* Wv = (const float*)d_in[6];
    const float* bv = (const float*)d_in[7];
    const float* Wo = (const float*)d_in[8];
    const float* bo = (const float*)d_in[9];
    const float* Wc = (const float*)d_in[10];
    const float* bc = (const float*)d_in[11];
    const float* Wg = (const float*)d_in[12];
    const float* bg = (const float*)d_in[13];

    // workspace layout
    char*   ws     = (char*)d_ws;
    size_t  vbytes = (size_t)B * D * sizeof(__bf16);       // 32 MiB each
    __bf16* v0     = (__bf16*)(ws + 0 * vbytes);
    __bf16* v1     = (__bf16*)(ws + 1 * vbytes);
    __bf16* v2     = (__bf16*)(ws + 2 * vbytes);
    __bf16* v3     = (__bf16*)(ws + 3 * vbytes);
    float*  att    = (float*) (ws + 4 * vbytes);           // 64 MiB
    __bf16* comb   = (__bf16*)(ws + 4 * vbytes + (size_t)B * D * sizeof(float)); // [B,2048]

    float*  fused  = (float*)d_out;                        // [B, D]
    float*  logits = (float*)d_out + (size_t)B * D;        // [B, 3]

    dim3 grid(D / TN, B / TM);   // (8, 128)
    dim3 block(256);

    // stage 1: V projections (f32 in -> bf16 panels)
    dia_gemm<EPI_STORE_BF16, false, 1024><<<grid, block, 0, stream>>>(drug_a, Wv + 0 * DD, bv + 0 * D, v0, D, 0, nullptr, D);
    dia_gemm<EPI_STORE_BF16, false, 1024><<<grid, block, 0, stream>>>(drug_b, Wv + 1 * DD, bv + 1 * D, v1, D, 0, nullptr, D);
    dia_gemm<EPI_STORE_BF16, false, 1024><<<grid, block, 0, stream>>>(drug_b, Wv + 2 * DD, bv + 2 * D, v2, D, 0, nullptr, D);
    dia_gemm<EPI_STORE_BF16, false, 1024><<<grid, block, 0, stream>>>(drug_a, Wv + 3 * DD, bv + 3 * D, v3, D, 0, nullptr, D);

    // stage 2: O projections; att = a_att + b_att (f32), cross panels packed [B,2048]
    dia_gemm<EPI_F32_INIT,   true, 1024><<<grid, block, 0, stream>>>(v0, Wo + 0 * DD, bo + 0 * D, att,  D,     0, nullptr, D);
    dia_gemm<EPI_F32_ACCUM,  true, 1024><<<grid, block, 0, stream>>>(v1, Wo + 1 * DD, bo + 1 * D, att,  D,     0, nullptr, D);
    dia_gemm<EPI_STORE_BF16, true, 1024><<<grid, block, 0, stream>>>(v2, Wo + 2 * DD, bo + 2 * D, comb, 2 * D, 0, nullptr, D);
    dia_gemm<EPI_STORE_BF16, true, 1024><<<grid, block, 0, stream>>>(v3, Wo + 3 * DD, bo + 3 * D, comb, 2 * D, D, nullptr, D);

    // stage 3: fused = att * sigmoid(comb @ Wg + bg)   (K = 2048)
    dia_gemm<EPI_GATE_FUSE,  true, 2048><<<grid, block, 0, stream>>>(comb, Wg, bg, fused, D, 0, att, 2 * D);

    // logits = comb @ Wc + bc
    dia_logits<<<B / 8, block, 0, stream>>>(comb, Wc, bc, logits, B);
}
